// CausalAutoformerBaseline_83940840833760
// MI455X (gfx1250) — matrix operand order repaired and checked by reading, not compiled, not run on hardware
//
#include <hip/hip_runtime.h>
#include <hip/hip_bf16.h>

// ---------------------------------------------------------------------------
// Autoformer encoder for MI455X (gfx1250, wave32, WMMA).
//
// All dense math runs through v_wmma_f32_16x16x32_bf16 (fp32 accumulate).
// The FFT auto-correlation is replaced by an exactly-equivalent Gram-matrix
// diagonal sum (corr mean over channels == circular diagonals of Q K^T),
// computed with WMMA and scattered into an LDS delay histogram.
//
// LDS operand tiles are laid out so every WMMA fragment is a contiguous
// 32-byte run per lane -> ds_load_b128; staging uses batched global_load_b128
// and packed ds_store_b64 writes.
//
// Workspace requirement: ~321 MB (see offsets in kernel_launch).
// ---------------------------------------------------------------------------

typedef __attribute__((ext_vector_type(16))) __bf16 v16bf;
typedef __attribute__((ext_vector_type(8)))  __bf16 v8bf;
typedef __attribute__((ext_vector_type(4)))  __bf16 v4bf;
typedef __attribute__((ext_vector_type(8)))  float  v8f;

#define AF_B   32
#define AF_L   2048
#define AF_D   128
#define AF_DFF 512
#define AF_TOPK 38
#define AF_S   (AF_B * AF_L)        // 65536 tokens
#define AF_LMASK (AF_L - 1)

static __device__ __forceinline__ float af_gelu(float x) {
    // jax.nn.gelu default (tanh approximation)
    float x3 = x * x * x;
    return 0.5f * x * (1.0f + tanhf(0.7978845608f * (x + 0.044715f * x3)));
}

// ---------------------------------------------------------------------------
// Embedding: h[token, c] = x[token, :8] @ W_in + b_in
// ---------------------------------------------------------------------------
__global__ __launch_bounds__(256) void af_embed(const float* __restrict__ x,
                                                const float* __restrict__ Wi,
                                                const float* __restrict__ bi,
                                                float* __restrict__ h) {
    size_t i = (size_t)blockIdx.x * 256 + threadIdx.x;   // over S*128
    int c = (int)(i & 127);
    size_t tok = i >> 7;
    float acc = bi[c];
#pragma unroll
    for (int j = 0; j < 8; ++j) acc += x[tok * 8 + j] * Wi[j * AF_D + c];
    h[i] = acc;
}

// ---------------------------------------------------------------------------
// Generic bf16 WMMA GEMM: C[M,N] = act(A[M,K] @ W[K,N] + bias[N])
// Block: 256 threads (8 waves). Tile: 64(M) x 128(N), K-step 32.
// Wave (mw in 0..1, nw in 0..3) computes a 2x2 grid of 16x16 subtiles
// (4 WMMA per k-step, A and B fragments each reused twice).
// sA is [m][k] (A fragment = two contiguous 16B runs); sBT is [n][k]
// (B fragment = one contiguous 32B run -> 2x ds_load_b128).
// Staging: all global float4 loads issued first (one load clause), then
// converted; B transpose uses 4x4 micro-tiles -> packed ds_store_b64.
// ---------------------------------------------------------------------------
__global__ __launch_bounds__(256) void af_gemm_bf16(const float* __restrict__ A,
                                                    const float* __restrict__ W,
                                                    const float* __restrict__ bias,
                                                    float* __restrict__ C,
                                                    int M, int K, int N, int act) {
    __shared__ __bf16 sA[64 * 32];    // [m][k]   4 KB
    __shared__ __bf16 sBT[128 * 32];  // [n][k]   8 KB

    const int tid  = threadIdx.x;
    const int m0   = blockIdx.x * 64;
    const int n0   = blockIdx.y * 128;
    const int wave = tid >> 5;
    const int lane = tid & 31;
    const int half = lane >> 4;       // lane half selects K/M sub-blocks
    const int mr   = lane & 15;
    const int nl   = lane & 15;
    const int mw   = wave >> 2;       // 0..1  (M subtile group: rows mw*32..mw*32+31)
    const int nw   = wave & 3;        // 0..3  (N subtile group: cols nw*32..nw*32+31)

    // Staging coordinates.
    // A: 2 float4 per thread, row-contiguous.
    const int aM0 = (tid * 4) >> 5,  aK0 = (tid * 4) & 31;          // i=0
    const int aM1 = ((1024 + tid * 4) >> 5), aK1 = (tid * 4) & 31;  // i=1
    // B: one 4(k) x 4(n) micro-tile per thread.
    const int bK = (tid >> 5) * 4;    // 0,4,...,28
    const int bN = (tid & 31) * 4;    // 0,4,...,124

    v8f acc00 = {}, acc01 = {}, acc10 = {}, acc11 = {};

    for (int k0 = 0; k0 < K; k0 += 32) {
        // --- issue all staging loads first (decoupled from waits) ---
        const float4 av0 = *(const float4*)&A[(size_t)(m0 + aM0) * K + (k0 + aK0)];
        const float4 av1 = *(const float4*)&A[(size_t)(m0 + aM1) * K + (k0 + aK1)];
        const float4 w0  = *(const float4*)&W[(size_t)(k0 + bK + 0) * N + (n0 + bN)];
        const float4 w1  = *(const float4*)&W[(size_t)(k0 + bK + 1) * N + (n0 + bN)];
        const float4 w2  = *(const float4*)&W[(size_t)(k0 + bK + 2) * N + (n0 + bN)];
        const float4 w3  = *(const float4*)&W[(size_t)(k0 + bK + 3) * N + (n0 + bN)];

        // Prefetch next k-step tiles into cache while we compute this one.
        if (k0 + 32 < K) {
            __builtin_prefetch(&A[(size_t)(m0 + (tid & 63)) * K + (k0 + 32)], 0, 3);
            __builtin_prefetch(&W[(size_t)(k0 + 32 + (tid >> 3)) * N + n0 + ((tid & 7) << 4)], 0, 3);
        }

        // --- convert + store to LDS ---
        // A rows: contiguous 4x bf16 -> ds_store_b64.
        {
            v4bf pa0 = { (__bf16)av0.x, (__bf16)av0.y, (__bf16)av0.z, (__bf16)av0.w };
            v4bf pa1 = { (__bf16)av1.x, (__bf16)av1.y, (__bf16)av1.z, (__bf16)av1.w };
            *(v4bf*)&sA[aM0 * 32 + aK0] = pa0;
            *(v4bf*)&sA[aM1 * 32 + aK1] = pa1;
        }
        // B transpose: pack 4 k-values per n-column -> ds_store_b64.
        {
            v4bf p0 = { (__bf16)w0.x, (__bf16)w1.x, (__bf16)w2.x, (__bf16)w3.x };
            v4bf p1 = { (__bf16)w0.y, (__bf16)w1.y, (__bf16)w2.y, (__bf16)w3.y };
            v4bf p2 = { (__bf16)w0.z, (__bf16)w1.z, (__bf16)w2.z, (__bf16)w3.z };
            v4bf p3 = { (__bf16)w0.w, (__bf16)w1.w, (__bf16)w2.w, (__bf16)w3.w };
            *(v4bf*)&sBT[(bN + 0) * 32 + bK] = p0;
            *(v4bf*)&sBT[(bN + 1) * 32 + bK] = p1;
            *(v4bf*)&sBT[(bN + 2) * 32 + bK] = p2;
            *(v4bf*)&sBT[(bN + 3) * 32 + bK] = p3;
        }
        __syncthreads();

        // --- fragments (contiguous k order, same for A and B, wave32) ---
        // A: element e -> K = e + 16*half  => two 16B runs.
        const int row0 = mw * 32 + mr;
        const int row1 = row0 + 16;
        v8bf a0lo = *(const v8bf*)&sA[row0 * 32 + 16 * half];
        v8bf a0hi = *(const v8bf*)&sA[row0 * 32 + 16 * half + 8];
        v8bf a1lo = *(const v8bf*)&sA[row1 * 32 + 16 * half];
        v8bf a1hi = *(const v8bf*)&sA[row1 * 32 + 16 * half + 8];
        v16bf af0 = __builtin_shufflevector(a0lo, a0hi, 0,1,2,3,4,5,6,7,8,9,10,11,12,13,14,15);
        v16bf af1 = __builtin_shufflevector(a1lo, a1hi, 0,1,2,3,4,5,6,7,8,9,10,11,12,13,14,15);
        // B: element e -> K = e + 16*half  => one 32B run per lane.
        const int nb0 = nw * 32, nb1 = nw * 32 + 16;
        v16bf bf0 = *(const v16bf*)&sBT[(nb0 + nl) * 32 + 16 * half];
        v16bf bf1 = *(const v16bf*)&sBT[(nb1 + nl) * 32 + 16 * half];

        acc00 = __builtin_amdgcn_wmma_f32_16x16x32_bf16(false, af0, false, bf0, (short)0, acc00, false, false);
        acc01 = __builtin_amdgcn_wmma_f32_16x16x32_bf16(false, af0, false, bf1, (short)0, acc01, false, false);
        acc10 = __builtin_amdgcn_wmma_f32_16x16x32_bf16(false, af1, false, bf0, (short)0, acc10, false, false);
        acc11 = __builtin_amdgcn_wmma_f32_16x16x32_bf16(false, af1, false, bf1, (short)0, acc11, false, false);
        __syncthreads();
    }

    // Epilogue: C/D layout -> row M = r + 8*half, col N = lane&15.
#pragma unroll
    for (int r = 0; r < 8; ++r) {
        int mmA = m0 + mw * 32 + r + 8 * half;
        int mmB = mmA + 16;
        int nn0 = n0 + nw * 32 + nl;
        int nn1 = nn0 + 16;
        float o00 = acc00[r] + bias[nn0];
        float o01 = acc01[r] + bias[nn1];
        float o10 = acc10[r] + bias[nn0];
        float o11 = acc11[r] + bias[nn1];
        if (act == 1) {
            o00 = af_gelu(o00); o01 = af_gelu(o01);
            o10 = af_gelu(o10); o11 = af_gelu(o11);
        }
        C[(size_t)mmA * N + nn0] = o00;
        C[(size_t)mmA * N + nn1] = o01;
        C[(size_t)mmB * N + nn0] = o10;
        C[(size_t)mmB * N + nn1] = o11;
    }
}

// ---------------------------------------------------------------------------
// Auto-correlation delay scores via Gram diagonals:
//   corr[b,d] += (1/128) * sum_t <q[b,t+d,:], k[b,t,:]>   (circular in t)
// Grid: (B, L/128). Block 256 (8 waves); wave w owns rows it0 = i0 + 16*w.
// K^T tile staged transposed (sKT[n][c]) so B fragments are contiguous 32B.
// Each 16x16 WMMA tile scatters into an LDS delay histogram at
// d = (i - j) mod L; histogram flushed with global atomics.
// ---------------------------------------------------------------------------
__global__ __launch_bounds__(256) void af_corr_gram(const float* __restrict__ q,
                                                    const float* __restrict__ k,
                                                    float* __restrict__ corr) {
    __shared__ __bf16 sKT[16 * 128];  // [n][c]   4 KB
    __shared__ float  sH[AF_L];       // delay histogram, 8 KB

    const int b    = blockIdx.x;
    const int i0   = blockIdx.y * 128;
    const int tid  = threadIdx.x;
    const int wave = tid >> 5;
    const int lane = tid & 31;
    const int half = lane >> 4;
    const int mr   = lane & 15;
    const int nl   = lane & 15;

    for (int i = tid; i < AF_L; i += 256) sH[i] = 0.0f;

    const float* qb = q + (size_t)b * AF_L * AF_D;
    const float* kb = k + (size_t)b * AF_L * AF_D;
    const int it0 = i0 + wave * 16;

    // Staging coordinates for the K^T tile (2 float4 per thread, [n][c]).
    const int kN0 = (tid * 4) >> 7,          kC0 = (tid * 4) & 127;
    const int kN1 = ((1024 + tid * 4) >> 7), kC1 = (tid * 4) & 127;

    // Preload the wave's A fragments (rows fixed, full K=128 -> 4 fragments).
    v16bf afrag[4];
    {
        const float* qrow = &qb[(size_t)(it0 + mr) * AF_D];
#pragma unroll
        for (int s = 0; s < 4; ++s) {
            const float4 q0 = *(const float4*)&qrow[s * 32 + 16 * half];
            const float4 q1 = *(const float4*)&qrow[s * 32 + 16 * half + 4];
            const float4 q2 = *(const float4*)&qrow[s * 32 + 16 * half + 8];
            const float4 q3 = *(const float4*)&qrow[s * 32 + 16 * half + 12];
            afrag[s][0]  = (__bf16)q0.x; afrag[s][1]  = (__bf16)q0.y;
            afrag[s][2]  = (__bf16)q0.z; afrag[s][3]  = (__bf16)q0.w;
            afrag[s][4]  = (__bf16)q1.x; afrag[s][5]  = (__bf16)q1.y;
            afrag[s][6]  = (__bf16)q1.z; afrag[s][7]  = (__bf16)q1.w;
            afrag[s][8]  = (__bf16)q2.x; afrag[s][9]  = (__bf16)q2.y;
            afrag[s][10] = (__bf16)q2.z; afrag[s][11] = (__bf16)q2.w;
            afrag[s][12] = (__bf16)q3.x; afrag[s][13] = (__bf16)q3.y;
            afrag[s][14] = (__bf16)q3.z; afrag[s][15] = (__bf16)q3.w;
        }
    }
    __syncthreads();

    for (int j0 = 0; j0 < AF_L; j0 += 16) {
        // Stage K^T tile: both float4 loads first, then convert + b64 stores.
        const float4 kv0 = *(const float4*)&kb[(size_t)(j0 + kN0) * AF_D + kC0];
        const float4 kv1 = *(const float4*)&kb[(size_t)(j0 + kN1) * AF_D + kC1];
        {
            v4bf p0 = { (__bf16)kv0.x, (__bf16)kv0.y, (__bf16)kv0.z, (__bf16)kv0.w };
            v4bf p1 = { (__bf16)kv1.x, (__bf16)kv1.y, (__bf16)kv1.z, (__bf16)kv1.w };
            *(v4bf*)&sKT[kN0 * 128 + kC0] = p0;
            *(v4bf*)&sKT[kN1 * 128 + kC1] = p1;
        }
        __syncthreads();

        v8f acc = {};
#pragma unroll
        for (int s = 0; s < 4; ++s) {
            // B fragment: element e -> channel s*32 + e + 16*half, contiguous.
            v16bf bfrag = *(const v16bf*)&sKT[nl * 128 + s * 32 + 16 * half];
            acc = __builtin_amdgcn_wmma_f32_16x16x32_bf16(false, afrag[s], false, bfrag,
                                                          (short)0, acc, false, false);
        }
        // Scatter G tile into delay histogram: d = (i - j) mod L.
#pragma unroll
        for (int r = 0; r < 8; ++r) {
            int ii = it0 + r + 8 * half;
            int jj = j0 + nl;
            int d  = (ii - jj) & AF_LMASK;
            atomicAdd(&sH[d], acc[r]);
        }
        __syncthreads();
    }

    for (int i = tid; i < AF_L; i += 256)
        atomicAdd(&corr[b * AF_L + i], sH[i] * (1.0f / 128.0f));
}

// corrB[d] = mean over batches of corr[b,d]
__global__ __launch_bounds__(256) void af_corr_bmean(const float* __restrict__ corr,
                                                     float* __restrict__ corrB) {
    int d = blockIdx.x * 256 + threadIdx.x;
    float s = 0.0f;
    for (int b = 0; b < AF_B; ++b) s += corr[b * AF_L + d];
    corrB[d] = s * (1.0f / AF_B);
}

// Top-K (K=38) delays over corrB: single block, iterative argmax in LDS.
__global__ __launch_bounds__(256) void af_topk(const float* __restrict__ corrB,
                                               int* __restrict__ idx) {
    __shared__ float sv[AF_L];
    __shared__ float rmax[256];
    __shared__ int   ridx[256];
    int tid = threadIdx.x;
    for (int i = tid; i < AF_L; i += 256) sv[i] = corrB[i];
    __syncthreads();
    for (int t = 0; t < AF_TOPK; ++t) {
        float bv = -3.4e38f; int bi = 0;
        for (int i = tid; i < AF_L; i += 256) {
            float v = sv[i];
            if (v > bv) { bv = v; bi = i; }
        }
        rmax[tid] = bv; ridx[tid] = bi;
        __syncthreads();
        for (int s = 128; s > 0; s >>= 1) {
            if (tid < s) {
                if (rmax[tid + s] > rmax[tid] ||
                    (rmax[tid + s] == rmax[tid] && ridx[tid + s] < ridx[tid])) {
                    rmax[tid] = rmax[tid + s];
                    ridx[tid] = ridx[tid + s];
                }
            }
            __syncthreads();
        }
        if (tid == 0) { idx[t] = ridx[0]; sv[ridx[0]] = -3.4e38f; }
        __syncthreads();
    }
}

// Per-batch softmax over the selected delays: w[b, 0..37]
__global__ __launch_bounds__(64) void af_softmax(const float* __restrict__ corr,
                                                 const int* __restrict__ idx,
                                                 float* __restrict__ w) {
    __shared__ float sv[AF_TOPK];
    __shared__ float ssum;
    int b = blockIdx.x, tid = threadIdx.x;
    if (tid < AF_TOPK) sv[tid] = corr[b * AF_L + idx[tid]];
    __syncthreads();
    if (tid == 0) {
        float m = -3.4e38f;
        for (int t = 0; t < AF_TOPK; ++t) m = fmaxf(m, sv[t]);
        float s = 0.0f;
        for (int t = 0; t < AF_TOPK; ++t) { sv[t] = __expf(sv[t] - m); s += sv[t]; }
        ssum = s;
    }
    __syncthreads();
    if (tid < AF_TOPK) w[b * AF_TOPK + tid] = sv[tid] / ssum;
}

// out[b,l,c4] = sum_t w[b,t] * v[b, (l + d_t) mod L, c4]   (float4 channels)
// Block: 128 threads = 4 tokens x 32 lanes; grid: S/4 blocks.
__global__ __launch_bounds__(128) void af_aggregate(const float* __restrict__ v,
                                                    const int* __restrict__ idx,
                                                    const float* __restrict__ w,
                                                    float* __restrict__ out) {
    __shared__ int   sd[AF_TOPK];
    __shared__ float sw[AF_TOPK];
    int tid = threadIdx.x;
    size_t tok = (size_t)blockIdx.x * 4 + (tid >> 5);
    int b = (int)(tok >> 11);
    int l = (int)(tok & AF_LMASK);
    int c4 = (tid & 31) * 4;
    if (tid < AF_TOPK) { sd[tid] = idx[tid]; sw[tid] = w[b * AF_TOPK + tid]; }
    __syncthreads();
    const float* vb = v + (size_t)b * AF_L * AF_D;
    float ax = 0.f, ay = 0.f, az = 0.f, aw = 0.f;
#pragma unroll 2
    for (int t = 0; t < AF_TOPK; ++t) {
        int ll = (l + sd[t]) & AF_LMASK;
        const float4 vv = *(const float4*)&vb[(size_t)ll * AF_D + c4];
        float wt = sw[t];
        ax += wt * vv.x; ay += wt * vv.y; az += wt * vv.z; aw += wt * vv.w;
    }
    float4 ov; ov.x = ax; ov.y = ay; ov.z = az; ov.w = aw;
    *(float4*)&out[tok * AF_D + c4] = ov;
}

// Elementwise residual add (float4)
__global__ __launch_bounds__(256) void af_add(const float* __restrict__ a,
                                              const float* __restrict__ b,
                                              float* __restrict__ o) {
    size_t i4 = (size_t)blockIdx.x * 256 + threadIdx.x;    // over S*32
    const float4 av = *(const float4*)&a[i4 * 4];
    const float4 bv = *(const float4*)&b[i4 * 4];
    float4 ov;
    ov.x = av.x + bv.x; ov.y = av.y + bv.y;
    ov.z = av.z + bv.z; ov.w = av.w + bv.w;
    *(float4*)&o[i4 * 4] = ov;
}

// series_decomp seasonal part: out = u - movavg_25(u), replicate padding (float4)
__global__ __launch_bounds__(256) void af_decomp(const float* __restrict__ u,
                                                 float* __restrict__ out) {
    size_t i4 = (size_t)blockIdx.x * 256 + threadIdx.x;    // over S*32
    int c4 = (int)(i4 & 31) * 4;
    size_t tok = i4 >> 5;
    int l = (int)(tok & AF_LMASK);
    size_t b = tok >> 11;
    const float* ub = u + (size_t)b * AF_L * AF_D;
    float sx = 0.f, sy = 0.f, sz = 0.f, sw = 0.f;
#pragma unroll
    for (int j = -12; j <= 12; ++j) {
        int lj = l + j;
        lj = lj < 0 ? 0 : (lj > AF_L - 1 ? AF_L - 1 : lj);
        const float4 t = *(const float4*)&ub[(size_t)lj * AF_D + c4];
        sx += t.x; sy += t.y; sz += t.z; sw += t.w;
    }
    const float4 uv = *(const float4*)&u[i4 * 4];
    float4 ov;
    ov.x = uv.x - sx * (1.0f / 25.0f);
    ov.y = uv.y - sy * (1.0f / 25.0f);
    ov.z = uv.z - sz * (1.0f / 25.0f);
    ov.w = uv.w - sw * (1.0f / 25.0f);
    *(float4*)&out[i4 * 4] = ov;
}

// my_Layernorm phase 1: per-token LN with gamma/beta
__global__ __launch_bounds__(128) void af_ln_token(const float* __restrict__ h,
                                                   const float* __restrict__ g,
                                                   const float* __restrict__ bta,
                                                   float* __restrict__ out) {
    __shared__ float red[128];
    size_t tok = blockIdx.x;
    int c = threadIdx.x;
    float x = h[tok * AF_D + c];
    red[c] = x; __syncthreads();
    for (int s = 64; s > 0; s >>= 1) { if (c < s) red[c] += red[c + s]; __syncthreads(); }
    float mu = red[0] * (1.0f / AF_D);
    __syncthreads();
    float dx = x - mu;
    red[c] = dx * dx; __syncthreads();
    for (int s = 64; s > 0; s >>= 1) { if (c < s) red[c] += red[c + s]; __syncthreads(); }
    float var = red[0] * (1.0f / AF_D);
    out[tok * AF_D + c] = dx * rsqrtf(var + 1e-5f) * g[c] + bta[c];
}

// my_Layernorm phase 2a: per (b,c) mean over L
__global__ __launch_bounds__(256) void af_colmean(const float* __restrict__ xh,
                                                  float* __restrict__ cm) {
    __shared__ float red[256];
    int bc = blockIdx.x;              // b*128 + c
    int b = bc >> 7, c = bc & 127;
    int tid = threadIdx.x;
    float s = 0.0f;
    for (int l = tid; l < AF_L; l += 256)
        s += xh[((size_t)b * AF_L + l) * AF_D + c];
    red[tid] = s; __syncthreads();
    for (int st = 128; st > 0; st >>= 1) { if (tid < st) red[tid] += red[tid + st]; __syncthreads(); }
    if (tid == 0) cm[bc] = red[0] * (1.0f / AF_L);
}

// my_Layernorm phase 2b: subtract seq-mean bias
__global__ __launch_bounds__(256) void af_subcol(const float* __restrict__ xh,
                                                 const float* __restrict__ cm,
                                                 float* __restrict__ out) {
    size_t i = (size_t)blockIdx.x * 256 + threadIdx.x;
    int c = (int)(i & 127);
    size_t tok = i >> 7;
    int b = (int)(tok >> 11);
    out[i] = xh[i] - cm[b * AF_D + c];
}

// Decode: out[tok, o] = h2[tok,:] @ W_dec + b_dec, o < 8
__global__ __launch_bounds__(256) void af_decode(const float* __restrict__ h2,
                                                 const float* __restrict__ Wd,
                                                 const float* __restrict__ bd,
                                                 float* __restrict__ out) {
    size_t i = (size_t)blockIdx.x * 256 + threadIdx.x;   // over S*8
    int o = (int)(i & 7);
    size_t tok = i >> 3;
    const float* hp = h2 + tok * AF_D;
    float acc = bd[o];
#pragma unroll 4
    for (int c = 0; c < AF_D; ++c) acc += hp[c] * Wd[c * 8 + o];
    out[i] = acc;
}

// ---------------------------------------------------------------------------
extern "C" void kernel_launch(void* const* d_in, const int* in_sizes, int n_in,
                              void* d_out, int out_size, void* d_ws, size_t ws_size,
                              hipStream_t stream) {
    (void)in_sizes; (void)n_in; (void)out_size; (void)ws_size;

    const float* x     = (const float*)d_in[0];
    const float* W_in  = (const float*)d_in[1];
    const float* b_in  = (const float*)d_in[2];
    const float* Wq    = (const float*)d_in[3];
    const float* bq    = (const float*)d_in[4];
    const float* Wk    = (const float*)d_in[5];
    const float* bk    = (const float*)d_in[6];
    const float* Wv    = (const float*)d_in[7];
    const float* bv    = (const float*)d_in[8];
    const float* Wo    = (const float*)d_in[9];
    const float* bo    = (const float*)d_in[10];
    const float* W1    = (const float*)d_in[11];
    const float* b1    = (const float*)d_in[12];
    const float* W2    = (const float*)d_in[13];
    const float* b2    = (const float*)d_in[14];
    const float* ln_g  = (const float*)d_in[15];
    const float* ln_b  = (const float*)d_in[16];
    const float* W_dec = (const float*)d_in[17];
    const float* b_dec = (const float*)d_in[18];
    float* out = (float*)d_out;

    // Workspace layout (bytes). Requires ~321 MB.
    char* ws = (char*)d_ws;
    const size_t MB = 1024ull * 1024ull;
    float* h    = (float*)(ws + 0 * MB);      // [S,128]   32 MB
    float* qb   = (float*)(ws + 32 * MB);     // [S,128]   32 MB (also reused for proj outs)
    float* kb   = (float*)(ws + 64 * MB);     // [S,128]   32 MB
    float* vb   = (float*)(ws + 96 * MB);     // [S,128]   32 MB
    float* ab   = (float*)(ws + 128 * MB);    // [S,128]   32 MB
    float* tb   = (float*)(ws + 160 * MB);    // [S,128]   32 MB
    float* ffb  = (float*)(ws + 192 * MB);    // [S,512]  128 MB
    char*  smal = ws + 320 * MB;
    float* corr    = (float*)(smal);                       // [B,L]   256 KB
    float* corrB   = (float*)(smal + 262144);              // [L]       8 KB
    int*   idx     = (int*)  (smal + 270336);              // [38]
    float* wts     = (float*)(smal + 270592);              // [B,38]
    float* colmean = (float*)(smal + 275456);              // [B,128]  16 KB

    const int S = AF_S;
    const int elem_blocks  = (S * AF_D) / 256;       // 32768 (scalar elementwise)
    const int elem4_blocks = (S * AF_D) / (256 * 4); // 8192  (float4 elementwise)
    dim3 gemm_grid_d(S / 64, 1);                // N = 128
    dim3 gemm_grid_ff(S / 64, AF_DFF / 128);    // N = 512

    af_embed<<<elem_blocks, 256, 0, stream>>>(x, W_in, b_in, h);

    for (int layer = 0; layer < 2; ++layer) {
        const float* wqL = Wq + (size_t)layer * AF_D * AF_D;
        const float* bqL = bq + (size_t)layer * AF_D;
        const float* wkL = Wk + (size_t)layer * AF_D * AF_D;
        const float* bkL = bk + (size_t)layer * AF_D;
        const float* wvL = Wv + (size_t)layer * AF_D * AF_D;
        const float* bvL = bv + (size_t)layer * AF_D;
        const float* woL = Wo + (size_t)layer * AF_D * AF_D;
        const float* boL = bo + (size_t)layer * AF_D;
        const float* w1L = W1 + (size_t)layer * AF_D * AF_DFF;
        const float* b1L = b1 + (size_t)layer * AF_DFF;
        const float* w2L = W2 + (size_t)layer * AF_DFF * AF_D;
        const float* b2L = b2 + (size_t)layer * AF_D;

        // QKV projections (bf16 WMMA, fp32 accumulate)
        af_gemm_bf16<<<gemm_grid_d, 256, 0, stream>>>(h, wqL, bqL, qb, S, AF_D, AF_D, 0);
        af_gemm_bf16<<<gemm_grid_d, 256, 0, stream>>>(h, wkL, bkL, kb, S, AF_D, AF_D, 0);
        af_gemm_bf16<<<gemm_grid_d, 256, 0, stream>>>(h, wvL, bvL, vb, S, AF_D, AF_D, 0);

        // Delay scores via WMMA Gram diagonals
        hipMemsetAsync(corr, 0, AF_B * AF_L * sizeof(float), stream);
        af_corr_gram<<<dim3(AF_B, AF_L / 128), 256, 0, stream>>>(qb, kb, corr);
        af_corr_bmean<<<AF_L / 256, 256, 0, stream>>>(corr, corrB);
        af_topk<<<1, 256, 0, stream>>>(corrB, idx);
        af_softmax<<<AF_B, 64, 0, stream>>>(corr, idx, wts);

        // Weighted rolled-V aggregation + output projection
        af_aggregate<<<S / 4, 128, 0, stream>>>(vb, idx, wts, ab);
        af_gemm_bf16<<<gemm_grid_d, 256, 0, stream>>>(ab, woL, boL, qb, S, AF_D, AF_D, 0);

        // x = decomp(h + attn)
        af_add<<<elem4_blocks, 256, 0, stream>>>(h, qb, tb);
        af_decomp<<<elem4_blocks, 256, 0, stream>>>(tb, h);

        // FFN: gelu(h@W1+b1)@W2+b2 ; x = decomp(h + ffn)
        af_gemm_bf16<<<gemm_grid_ff, 256, 0, stream>>>(h, w1L, b1L, ffb, S, AF_D, AF_DFF, 1);
        af_gemm_bf16<<<gemm_grid_d, 256, 0, stream>>>(ffb, w2L, b2L, qb, S, AF_DFF, AF_D, 0);
        af_add<<<elem4_blocks, 256, 0, stream>>>(h, qb, tb);
        af_decomp<<<elem4_blocks, 256, 0, stream>>>(tb, h);
    }

    // my_Layernorm + decode
    af_ln_token<<<S, 128, 0, stream>>>(h, ln_g, ln_b, ab);
    af_colmean<<<AF_B * AF_D, 256, 0, stream>>>(ab, colmean);
    af_subcol<<<elem_blocks, 256, 0, stream>>>(ab, colmean, tb);
    af_decode<<<(S * 8) / 256, 256, 0, stream>>>(tb, W_dec, b_dec, out);
}
